// _RWKV6Block_72206990180960
// MI455X (gfx1250) — compile-verified, matrix operand order checked by reading, not run on hardware
//
#include <hip/hip_runtime.h>
#include <hip/hip_bf16.h>
#include <math.h>

typedef __attribute__((ext_vector_type(16))) _Float16 v16h;
typedef __attribute__((ext_vector_type(8)))  _Float16 v8h;
typedef __attribute__((ext_vector_type(8)))  float    v8f;

#define LN_EPS 1e-5f
#define Bsz 4
#define Tsz 4096
#define Csz 1024
#define Msz (Bsz * Tsz)     // 16384 rows
#define NCHUNK 16
#define CLEN (Tsz / NCHUNK) // 256

__device__ __forceinline__ float sigmoidf_(float v) {
    return 1.f / (1.f + __expf(-v));
}

// ---------------------------------------------------------------------------
// LayerNorm: one block (256 threads) per row of C=1024 floats.
// ---------------------------------------------------------------------------
__global__ __launch_bounds__(256) void k_layernorm(
    const float* __restrict__ x, const float* __restrict__ w,
    const float* __restrict__ b, float* __restrict__ xt)
{
    const int row = blockIdx.x;
    const float* xr = x + (size_t)row * Csz;
    float s = 0.f, s2 = 0.f;
    for (int c = threadIdx.x; c < Csz; c += 256) {
        float v = xr[c];
        s += v; s2 += v * v;
    }
    for (int off = 16; off; off >>= 1) {
        s  += __shfl_xor(s,  off);
        s2 += __shfl_xor(s2, off);
    }
    __shared__ float ssum[8], ssum2[8];
    const int lane = threadIdx.x & 31, wv = threadIdx.x >> 5;
    if (lane == 0) { ssum[wv] = s; ssum2[wv] = s2; }
    __syncthreads();
    float ts = 0.f, ts2 = 0.f;
    #pragma unroll
    for (int i = 0; i < 8; ++i) { ts += ssum[i]; ts2 += ssum2[i]; }
    const float mu   = ts / (float)Csz;
    const float var  = ts2 / (float)Csz - mu * mu;
    const float rstd = rsqrtf(var + LN_EPS);
    float* o = xt + (size_t)row * Csz;
    for (int c = threadIdx.x; c < Csz; c += 256)
        o[c] = (xr[c] - mu) * rstd * w[c] + b[c];
}

// ---------------------------------------------------------------------------
// Token mix: xk/xv/xr = xt*m + x_prev*(1-m), stored f16 for the WMMA GEMMs.
// ---------------------------------------------------------------------------
__global__ __launch_bounds__(256) void k_mix(
    const float* __restrict__ xt,
    const float* __restrict__ mk, const float* __restrict__ mv,
    const float* __restrict__ mr,
    _Float16* __restrict__ xk, _Float16* __restrict__ xv,
    _Float16* __restrict__ xr)
{
    const size_t i = (size_t)blockIdx.x * 256 + threadIdx.x;
    const int c   = (int)(i & (Csz - 1));
    const int row = (int)(i >> 10);
    const int t   = row & (Tsz - 1);
    const float cur  = xt[i];
    const float prev = (t == 0) ? 0.f : xt[i - Csz];
    const float k = mk[c], v = mv[c], r = mr[c];
    xk[i] = (_Float16)(cur * k + prev * (1.f - k));
    xv[i] = (_Float16)(cur * v + prev * (1.f - v));
    xr[i] = (_Float16)(cur * r + prev * (1.f - r));
}

// ---------------------------------------------------------------------------
// Weight convert f32 -> f16 (all four matrices in one pass).
// ---------------------------------------------------------------------------
__global__ __launch_bounds__(256) void k_cvtw(
    const float* __restrict__ wk, const float* __restrict__ wv,
    const float* __restrict__ wr, const float* __restrict__ wo,
    _Float16* __restrict__ hk, _Float16* __restrict__ hv,
    _Float16* __restrict__ hr, _Float16* __restrict__ ho)
{
    const size_t i = (size_t)blockIdx.x * 256 + threadIdx.x;
    hk[i] = (_Float16)wk[i];
    hv[i] = (_Float16)wv[i];
    hr[i] = (_Float16)wr[i];
    ho[i] = (_Float16)wo[i];
}

// ---------------------------------------------------------------------------
// Async global -> LDS copy of one 16-byte chunk (CDNA5 ASYNCcnt-tracked).
// GVS mode: saddr = 64-bit SGPR base, vaddr = 32-bit byte offset,
// vdst VGPR holds the LDS byte address.
// ---------------------------------------------------------------------------
__device__ __forceinline__ void async_b128_to_lds(unsigned lds_byte_off,
                                                  const void* gbase,
                                                  unsigned gbyte_off)
{
    asm volatile("global_load_async_to_lds_b128 %0, %1, %2"
                 :: "v"(lds_byte_off), "v"(gbyte_off), "s"(gbase)
                 : "memory");
}
__device__ __forceinline__ void wait_asynccnt0()
{
    asm volatile("s_wait_asynccnt 0x0" ::: "memory");
}

// ---------------------------------------------------------------------------
// WMMA GEMM:  Out[m,n] = sum_k A[m,k] * W[n,k]   (A: [M,K] f16, W: [N,K] f16)
// Block: 256 threads / 8 waves -> 256x64 output tile.
// Wave:  32x64 strip = 2x4 accumulators, 8 WMMAs per K-step of 32.
// B tile (64x32 f16, 4 KB) staged in LDS via double-buffered async copies;
// K-loop unrolled x2 so the buffer parity is static and the prefetch is
// branch-free (final step re-prefetches k=0 into the dead buffer).
// A fragments stream straight from global (rows are wave-private, b128).
// Fragment layout per CDNA5 ISA 16-bit A 16x32: lanes 0-15 hold K 0-7/16-23,
// lanes 16-31 hold K 8-15/24-31; B columns loaded identically from W rows.
// EPI: 0 = f32 store (K), 1 = f16 (V), 2 = sigmoid->f16 (R), 3 = +resid f32.
// ---------------------------------------------------------------------------
template <int EPI>
__global__ __launch_bounds__(256) void k_gemm_wmma(
    const _Float16* __restrict__ A, const _Float16* __restrict__ W,
    const float* __restrict__ resid, void* __restrict__ Out,
    int M, int N, int K)
{
    __shared__ _Float16 smem[2 * 64 * 32];   // 8 KB, two B-tile buffers

    const int lane = threadIdx.x & 31;
    const int wave = threadIdx.x >> 5;
    const int hi   = lane >> 4;
    const int l15  = lane & 15;
    const int tm   = blockIdx.y * 256 + wave * 32;
    const int tn   = blockIdx.x * 64;

    // cooperative copy indices: thread -> (row n of 64, 16B segment of 4)
    const int cp_n   = threadIdx.x >> 2;
    const int cp_seg = threadIdx.x & 3;
    const unsigned cp_lds = (unsigned)(cp_n * 64 + cp_seg * 16);
    const unsigned cp_gb  = (unsigned)(((tn + cp_n) * K + cp_seg * 8) * 2);

    v8f acc[2][4];
    #pragma unroll
    for (int s = 0; s < 2; ++s)
        #pragma unroll
        for (int j = 0; j < 4; ++j) acc[s][j] = (v8f){};

    const _Float16* arow0 = A + (size_t)(tm +  0 + l15) * K;
    const _Float16* arow1 = A + (size_t)(tm + 16 + l15) * K;

    // one 32-wide K-step against LDS buffer `p` (compile-time constant)
    auto kstep = [&](int k0, int pbuf, int knext) {
        wait_asynccnt0();          // our buffer-p writes complete
        __syncthreads();           // everyone's writes done, prior reads done
        // branch-free prefetch of the next tile into the other buffer
        async_b128_to_lds((unsigned)((1 - pbuf) * 4096) + cp_lds, W,
                          cp_gb + (unsigned)(knext * 2));

        const int ka = k0 + hi * 8;
        v16h af0, af1, bf[4];
        {
            v8h lo = *(const v8h*)(arow0 + ka);
            v8h hp = *(const v8h*)(arow0 + ka + 16);
            #pragma unroll
            for (int i = 0; i < 8; ++i) { af0[i] = lo[i]; af0[8 + i] = hp[i]; }
        }
        {
            v8h lo = *(const v8h*)(arow1 + ka);
            v8h hp = *(const v8h*)(arow1 + ka + 16);
            #pragma unroll
            for (int i = 0; i < 8; ++i) { af1[i] = lo[i]; af1[8 + i] = hp[i]; }
        }
        #pragma unroll
        for (int j = 0; j < 4; ++j) {   // batch all B-fragment DS loads first
            const _Float16* bb =
                &smem[pbuf * 2048 + (j * 16 + l15) * 32 + hi * 8];
            v8h lo = *(const v8h*)(bb);
            v8h hp = *(const v8h*)(bb + 16);
            #pragma unroll
            for (int i = 0; i < 8; ++i) { bf[j][i] = lo[i]; bf[j][8 + i] = hp[i]; }
        }
        #pragma unroll
        for (int j = 0; j < 4; ++j) {
            acc[0][j] = __builtin_amdgcn_wmma_f32_16x16x32_f16(
                false, af0, false, bf[j], (short)0, acc[0][j], false, false);
            acc[1][j] = __builtin_amdgcn_wmma_f32_16x16x32_f16(
                false, af1, false, bf[j], (short)0, acc[1][j], false, false);
        }
    };

    // prologue: stage first B tile into buffer 0
    async_b128_to_lds(cp_lds, W, cp_gb);

    for (int k0 = 0; k0 < K; k0 += 64) {
        const int kn2 = (k0 + 64 < K) ? (k0 + 64) : 0;  // wrap: dead prefetch
        kstep(k0,      0, k0 + 32);
        kstep(k0 + 32, 1, kn2);
    }

    // C/D layout: VGPR r, lanes 0-15 -> (M=r, N=lane); lanes 16-31 -> (M=r+8).
    #pragma unroll
    for (int s = 0; s < 2; ++s) {
        const int rbase = tm + s * 16 + hi * 8;
        #pragma unroll
        for (int j = 0; j < 4; ++j) {
            const int col = tn + j * 16 + l15;
            #pragma unroll
            for (int r = 0; r < 8; ++r) {
                const size_t o = (size_t)(rbase + r) * N + col;
                const float v = acc[s][j][r];
                if (EPI == 0) {
                    ((float*)Out)[o] = v;
                } else if (EPI == 1) {
                    ((_Float16*)Out)[o] = (_Float16)v;
                } else if (EPI == 2) {
                    ((_Float16*)Out)[o] = (_Float16)sigmoidf_(v);
                } else {
                    ((float*)Out)[o] = resid[o] + v;
                }
            }
        }
    }
}

// ---------------------------------------------------------------------------
// Chunked linear recurrence (16 chunks of 256 steps -> 64K-thread passes).
// Pass 1: chunk-local end states (zero incoming).
// ---------------------------------------------------------------------------
__global__ __launch_bounds__(256) void k_scan_local(
    const float* __restrict__ Kf, const float* __restrict__ tdecay,
    float* __restrict__ cstate)
{
    const int tid   = blockIdx.x * 256 + threadIdx.x;   // b*NCHUNK*C
    const int c     = tid & (Csz - 1);
    const int bc    = tid >> 10;
    const int chunk = bc & (NCHUNK - 1);
    const int b     = bc >> 4;
    const float decay = sigmoidf_(tdecay[c]);
    float st = 0.f;
    size_t idx = ((size_t)b * Tsz + (size_t)chunk * CLEN) * Csz + c;
    for (int t = 0; t < CLEN; ++t, idx += Csz)
        st = st * decay + Kf[idx];
    cstate[tid] = st;
}

// Pass 2: serial propagation across the 16 chunks per (b,c); converts
// chunk-end states (in place) into chunk-incoming states.
__global__ __launch_bounds__(256) void k_scan_prop(
    const float* __restrict__ tdecay, float* __restrict__ cstate)
{
    const int tid = blockIdx.x * 256 + threadIdx.x;     // b*C
    const int c   = tid & (Csz - 1);
    const int b   = tid >> 10;
    const float decay = sigmoidf_(tdecay[c]);
    const float dL    = __powf(decay, (float)CLEN);
    float s = 0.f;
    #pragma unroll
    for (int j = 0; j < NCHUNK; ++j) {
        const size_t o = ((size_t)b * NCHUNK + j) * Csz + c;
        const float e = cstate[o];
        cstate[o] = s;            // incoming state for chunk j
        s = s * dL + e;
    }
}

// Pass 3: recompute the scan with the true incoming state and emit the
// fused epilogue  a = R * ((S*first + K) * V)  as the f16 A-operand.
__global__ __launch_bounds__(256) void k_scan_apply(
    const float* __restrict__ Kf, const _Float16* __restrict__ Vh,
    const _Float16* __restrict__ Rh, const float* __restrict__ tdecay,
    const float* __restrict__ tfirst, const float* __restrict__ cstate,
    _Float16* __restrict__ Af)
{
    const int tid   = blockIdx.x * 256 + threadIdx.x;   // b*NCHUNK*C
    const int c     = tid & (Csz - 1);
    const int bc    = tid >> 10;
    const int chunk = bc & (NCHUNK - 1);
    const int b     = bc >> 4;
    const float decay = sigmoidf_(tdecay[c]);
    const float first = __expf(tfirst[c]);
    float st = cstate[tid];
    size_t idx = ((size_t)b * Tsz + (size_t)chunk * CLEN) * Csz + c;
    for (int t = 0; t < CLEN; ++t, idx += Csz) {
        const float k = Kf[idx];
        st = st * decay + k;
        const float v = (float)Vh[idx];
        const float r = (float)Rh[idx];
        Af[idx] = (_Float16)(r * ((st * first + k) * v));
    }
}

// ---------------------------------------------------------------------------
// Launch
// ---------------------------------------------------------------------------
extern "C" void kernel_launch(void* const* d_in, const int* in_sizes, int n_in,
                              void* d_out, int out_size, void* d_ws, size_t ws_size,
                              hipStream_t stream)
{
    (void)in_sizes; (void)n_in; (void)out_size; (void)ws_size;
    const float* x      = (const float*)d_in[0];
    const float* mk     = (const float*)d_in[1];
    const float* mv     = (const float*)d_in[2];
    const float* mr     = (const float*)d_in[3];
    const float* tdecay = (const float*)d_in[4];
    const float* tfirst = (const float*)d_in[5];
    const float* Wk     = (const float*)d_in[6];
    const float* Wv     = (const float*)d_in[7];
    const float* Wr     = (const float*)d_in[8];
    const float* Wo     = (const float*)d_in[9];
    const float* lnw    = (const float*)d_in[10];
    const float* lnb    = (const float*)d_in[11];
    float* out = (float*)d_out;

    char* p = (char*)d_ws;
    auto take = [&](size_t bytes) {
        char* r = p;
        p += (bytes + 255) & ~(size_t)255;
        return (void*)r;
    };
    const size_t MC = (size_t)Msz * Csz;
    const size_t CC = (size_t)Csz * Csz;
    float*    xt  = (float*)   take(MC * 4);
    _Float16* xk  = (_Float16*)take(MC * 2);
    _Float16* xv  = (_Float16*)take(MC * 2);
    _Float16* xr  = (_Float16*)take(MC * 2);
    _Float16* hWk = (_Float16*)take(CC * 2);
    _Float16* hWv = (_Float16*)take(CC * 2);
    _Float16* hWr = (_Float16*)take(CC * 2);
    _Float16* hWo = (_Float16*)take(CC * 2);
    float*    Kf  = (float*)   take(MC * 4);
    _Float16* Vh  = (_Float16*)take(MC * 2);
    _Float16* Rh  = (_Float16*)take(MC * 2);
    _Float16* Af  = (_Float16*)take(MC * 2);
    float*    cst = (float*)   take((size_t)Bsz * NCHUNK * Csz * 4);

    k_cvtw<<<(int)(CC / 256), 256, 0, stream>>>(Wk, Wv, Wr, Wo, hWk, hWv, hWr, hWo);
    k_layernorm<<<Msz, 256, 0, stream>>>(x, lnw, lnb, xt);
    k_mix<<<(int)(MC / 256), 256, 0, stream>>>(xt, mk, mv, mr, xk, xv, xr);

    dim3 gg(Csz / 64, Msz / 256);
    k_gemm_wmma<0><<<gg, 256, 0, stream>>>(xk, hWk, nullptr, (void*)Kf, Msz, Csz, Csz);
    k_gemm_wmma<1><<<gg, 256, 0, stream>>>(xv, hWv, nullptr, (void*)Vh, Msz, Csz, Csz);
    k_gemm_wmma<2><<<gg, 256, 0, stream>>>(xr, hWr, nullptr, (void*)Rh, Msz, Csz, Csz);

    k_scan_local<<<(Bsz * NCHUNK * Csz) / 256, 256, 0, stream>>>(Kf, tdecay, cst);
    k_scan_prop <<<(Bsz * Csz) / 256, 256, 0, stream>>>(tdecay, cst);
    k_scan_apply<<<(Bsz * NCHUNK * Csz) / 256, 256, 0, stream>>>(Kf, Vh, Rh, tdecay,
                                                                 tfirst, cst, Af);

    k_gemm_wmma<3><<<gg, 256, 0, stream>>>(Af, hWo, x, (void*)out, Msz, Csz, Csz);
}
